// SelfAttention_28930899706122
// MI455X (gfx1250) — matrix operand order repaired and checked by reading, not compiled, not run on hardware
//
#include <hip/hip_runtime.h>
#include <hip/hip_bf16.h>
#include <stdint.h>

// ---------------------------------------------------------------------------
// Problem constants (reference: B=8, S=1024, D=1024, H=16, DK=DV=64)
// ---------------------------------------------------------------------------
#define B_   8
#define S_   1024
#define D_   1024
#define H_   16
#define DK_  64
#define NROWS (B_ * S_)   // 8192 token rows

typedef __attribute__((ext_vector_type(16))) __bf16 v16bf;
typedef __attribute__((ext_vector_type(8)))  __bf16 v8bf;
typedef __attribute__((ext_vector_type(8)))  float  v8f;

// float -> bf16 bits, round-to-nearest-even
__device__ __forceinline__ unsigned short f2bf(float f) {
  unsigned int u = __float_as_uint(f);
  u += 0x7FFFu + ((u >> 16) & 1u);
  return (unsigned short)(u >> 16);
}

// A-fragment (16x32 MxK, 16-bit): lane<16 holds K[0..15]; lane>=16 K[16..31]
// p points at (row base + kc); each half is one 16-byte load.
__device__ __forceinline__ v16bf load_a_frag(const unsigned short* p, bool hi) {
  const __bf16* q = (const __bf16*)p;
  union { v16bf v; v8bf h[2]; } u;
  const int off = hi ? 16 : 0;
  u.h[0] = *(const v8bf*)(q + off);
  u.h[1] = *(const v8bf*)(q + off + 8);
  return u.v;
}
// B-fragment (32x16 KxN, 16-bit): lane<16 holds K[0..15]; lane>=16 K[16..31] (column = lane&15)
__device__ __forceinline__ v16bf load_b_frag(const unsigned short* p, bool hi) {
  const __bf16* q = (const __bf16*)p;
  union { v16bf v; v8bf h[2]; } u;
  const int off = hi ? 16 : 0;
  u.h[0] = *(const v8bf*)(q + off);
  u.h[1] = *(const v8bf*)(q + off + 8);
  return u.v;
}

__device__ __forceinline__ v8f wmma_bf16(v16bf a, v16bf b, v8f c) {
  // D = A*B + C, fp32 accumulate  -> v_wmma_f32_16x16x32_bf16
  return __builtin_amdgcn_wmma_f32_16x16x32_bf16(false, a, false, b, (short)0, c,
                                                 false, false);
}

// ---------------------------------------------------------------------------
// Kernel 1: fp32 -> bf16 weight conversion
// ---------------------------------------------------------------------------
__global__ void cvt_kernel(const float* __restrict__ src,
                           unsigned short* __restrict__ dst, int n) {
  for (int i = blockIdx.x * blockDim.x + threadIdx.x; i < n;
       i += gridDim.x * blockDim.x)
    dst[i] = f2bf(src[i]);
}

// ---------------------------------------------------------------------------
// Kernel 2: LayerNorm (pre-LN) + cast to bf16.  One 256-thread block per row.
// ---------------------------------------------------------------------------
__global__ __launch_bounds__(256) void ln_kernel(const float* __restrict__ x,
                                                 const float* __restrict__ g,
                                                 const float* __restrict__ beta,
                                                 unsigned short* __restrict__ xn) {
  __shared__ float red[256];
  const int tid = threadIdx.x;
  const size_t row = blockIdx.x;
  const float* xr = x + row * D_;
  float v0[4];
  float s = 0.f, s2 = 0.f;
#pragma unroll
  for (int i = 0; i < 4; ++i) {
    float v = xr[tid + i * 256];
    v0[i] = v; s += v; s2 += v * v;
  }
  red[tid] = s; __syncthreads();
  for (int o = 128; o > 0; o >>= 1) { if (tid < o) red[tid] += red[tid + o]; __syncthreads(); }
  const float mean = red[0] * (1.f / D_);
  __syncthreads();
  red[tid] = s2; __syncthreads();
  for (int o = 128; o > 0; o >>= 1) { if (tid < o) red[tid] += red[tid + o]; __syncthreads(); }
  const float var = red[0] * (1.f / D_) - mean * mean;
  const float rstd = rsqrtf(var + 1e-6f);
#pragma unroll
  for (int i = 0; i < 4; ++i) {
    const int c = tid + i * 256;
    xn[row * D_ + c] = f2bf((v0[i] - mean) * rstd * g[c] + beta[c]);
  }
}

// ---------------------------------------------------------------------------
// Kernel 3: QKV projection GEMM (NT): [8192,1024] x [1024,1024]^T, bf16 WMMA.
// blockIdx.z selects Q/K/V.  Block = 4 waves, each wave -> 16x64 tile.
// Q,K written [B,H,S,64]; V written transposed [B,H,64,S] for P·V fragments.
// ---------------------------------------------------------------------------
__global__ __launch_bounds__(128) void proj_kernel(
    const unsigned short* __restrict__ xn,
    const unsigned short* __restrict__ wq,
    const unsigned short* __restrict__ wk,
    const unsigned short* __restrict__ wv,
    unsigned short* __restrict__ Qb,
    unsigned short* __restrict__ Kb,
    unsigned short* __restrict__ Vt) {
  const int lane = threadIdx.x & 31;
  const int wave = threadIdx.x >> 5;
  const bool hi = lane >= 16;
  const int ln16 = lane & 15;
  const int mode = blockIdx.z;
  const unsigned short* W = (mode == 0) ? wq : (mode == 1) ? wk : wv;

  const int m0 = blockIdx.x * 64 + wave * 16;
  const int n0 = blockIdx.y * 64;

  v8f acc[4] = {};
  const unsigned short* arow = xn + (size_t)(m0 + ln16) * D_;
  for (int kc = 0; kc < D_; kc += 32) {
    const v16bf a = load_a_frag(arow + kc, hi);
#pragma unroll
    for (int nt = 0; nt < 4; ++nt) {
      const v16bf b = load_b_frag(W + (size_t)(n0 + nt * 16 + ln16) * D_ + kc, hi);
      acc[nt] = wmma_bf16(a, b, acc[nt]);
    }
  }
#pragma unroll
  for (int nt = 0; nt < 4; ++nt) {
    const int n = n0 + nt * 16 + ln16;
    const int h = n >> 6, d = n & 63;
#pragma unroll
    for (int v = 0; v < 8; ++v) {
      const int m = m0 + v + (hi ? 8 : 0);
      const int bb = m >> 10, s = m & (S_ - 1);
      const unsigned short val = f2bf(acc[nt][v]);
      if (mode == 0)
        Qb[(((size_t)(bb * H_ + h)) * S_ + s) * DK_ + d] = val;
      else if (mode == 1)
        Kb[(((size_t)(bb * H_ + h)) * S_ + s) * DK_ + d] = val;
      else
        Vt[(((size_t)(bb * H_ + h)) * DK_ + d) * S_ + s] = val;
    }
  }
}

// ---------------------------------------------------------------------------
// Kernel 4: attention.  Block = 8 waves, one (b,h, 16-query tile) per block.
// Wave w owns kv columns [w*128, w*128+128).
// ---------------------------------------------------------------------------
__global__ __launch_bounds__(256) void attn_kernel(
    const unsigned short* __restrict__ Qb,
    const unsigned short* __restrict__ Kb,
    const unsigned short* __restrict__ Vt,
    const unsigned char* __restrict__ mask,
    float* __restrict__ attn_out,
    unsigned short* __restrict__ ctx) {
  __shared__ __attribute__((aligned(16))) unsigned short sP[16 * S_];  // 32 KB
  __shared__ float sStat[8][16];
  __shared__ float sRow[16];
  __shared__ float sInv[16];

  const int lane = threadIdx.x & 31;
  const int wave = threadIdx.x >> 5;
  const bool hi = lane >= 16;
  const int ln16 = lane & 15;
  const int bh = blockIdx.y;               // b*H + h
  const int b = bh >> 4, h = bh & 15;
  const int q0 = blockIdx.x * 16;
  const size_t qbase = ((size_t)bh * S_ + q0) * DK_;
  const int kv0 = wave * 128;

  // Q fragments (16x64 tile -> two K=32 fragments)
  const v16bf aq0 = load_a_frag(Qb + qbase + (size_t)ln16 * DK_, hi);
  const v16bf aq1 = load_a_frag(Qb + qbase + (size_t)ln16 * DK_ + 32, hi);

  // scores: 8 tiles of 16(q) x 16(kv) per wave
  v8f sc[8];
#pragma unroll
  for (int t = 0; t < 8; ++t) {
    const int kv = kv0 + t * 16 + ln16;
    const unsigned short* kp = Kb + ((size_t)bh * S_ + kv) * DK_;
    v8f a = {};
    a = wmma_bf16(aq0, load_b_frag(kp, hi), a);
    a = wmma_bf16(aq1, load_b_frag(kp + 32, hi), a);
    sc[t] = a;
  }

  // scale 1/sqrt(64) + mask
#pragma unroll
  for (int t = 0; t < 8; ++t) {
    const int kv = kv0 + t * 16 + ln16;
#pragma unroll
    for (int v = 0; v < 8; ++v) {
      const int m = v + (hi ? 8 : 0);
      float s = sc[t][v] * 0.125f;
      if (mask[((size_t)b * S_ + (q0 + m)) * S_ + kv]) s = -1e9f;
      sc[t][v] = s;
    }
  }

  // row max: per-lane over tiles, shuffle over the 16 columns, LDS over waves
  float pr[8];
#pragma unroll
  for (int v = 0; v < 8; ++v) {
    float mx = sc[0][v];
#pragma unroll
    for (int t = 1; t < 8; ++t) mx = fmaxf(mx, sc[t][v]);
    pr[v] = mx;
  }
  for (int off = 1; off < 16; off <<= 1) {
#pragma unroll
    for (int v = 0; v < 8; ++v) pr[v] = fmaxf(pr[v], __shfl_xor(pr[v], off, 32));
  }
  if (lane == 0)  for (int v = 0; v < 8; ++v) sStat[wave][v] = pr[v];
  if (lane == 16) for (int v = 0; v < 8; ++v) sStat[wave][v + 8] = pr[v];
  __syncthreads();
  if (threadIdx.x < 16) {
    float mx = sStat[0][threadIdx.x];
    for (int w = 1; w < 8; ++w) mx = fmaxf(mx, sStat[w][threadIdx.x]);
    sRow[threadIdx.x] = mx;
  }
  __syncthreads();

  // exp(s - rowmax), stage bf16 P in LDS, accumulate row sums
#pragma unroll
  for (int v = 0; v < 8; ++v) pr[v] = 0.f;
#pragma unroll
  for (int t = 0; t < 8; ++t) {
#pragma unroll
    for (int v = 0; v < 8; ++v) {
      const int m = v + (hi ? 8 : 0);
      const float p = __expf(sc[t][v] - sRow[m]);
      sc[t][v] = p;
      pr[v] += p;
      sP[m * S_ + kv0 + t * 16 + ln16] = f2bf(p);
    }
  }
  for (int off = 1; off < 16; off <<= 1) {
#pragma unroll
    for (int v = 0; v < 8; ++v) pr[v] += __shfl_xor(pr[v], off, 32);
  }
  if (lane == 0)  for (int v = 0; v < 8; ++v) sStat[wave][v] = pr[v];
  if (lane == 16) for (int v = 0; v < 8; ++v) sStat[wave][v + 8] = pr[v];
  __syncthreads();
  if (threadIdx.x < 16) {
    float sm = 0.f;
    for (int w = 0; w < 8; ++w) sm += sStat[w][threadIdx.x];
    sInv[threadIdx.x] = 1.f / sm;
  }
  __syncthreads();

  // write normalized attention probabilities (fp32, [B,H,S,S])
#pragma unroll
  for (int t = 0; t < 8; ++t) {
    const int kv = kv0 + t * 16 + ln16;
#pragma unroll
    for (int v = 0; v < 8; ++v) {
      const int m = v + (hi ? 8 : 0);
      attn_out[((size_t)bh * S_ + (q0 + m)) * S_ + kv] = sc[t][v] * sInv[m];
    }
  }

  // ctx = P @ V  : K split across waves; A from LDS (ds_load_b128), B from Vt
  v8f cacc[4] = {};
  for (int kc = kv0; kc < kv0 + 128; kc += 32) {
    const v16bf a = load_a_frag(&sP[ln16 * S_ + kc], hi);
#pragma unroll
    for (int nt = 0; nt < 4; ++nt) {
      const int d = nt * 16 + ln16;
      const v16bf bv = load_b_frag(Vt + ((size_t)bh * DK_ + d) * S_ + kc, hi);
      cacc[nt] = wmma_bf16(a, bv, cacc[nt]);
    }
  }
  __syncthreads();  // everyone done reading sP -> reuse as fp32 reduce buffer
  float* sCtx = (float*)sP;  // 8 waves * 16 * 64 floats = 32 KB exactly
#pragma unroll
  for (int nt = 0; nt < 4; ++nt) {
    const int d = nt * 16 + ln16;
#pragma unroll
    for (int v = 0; v < 8; ++v) {
      const int m = v + (hi ? 8 : 0);
      sCtx[(wave * 16 + m) * DK_ + d] = cacc[nt][v];
    }
  }
  __syncthreads();
  for (int i = threadIdx.x; i < 16 * DK_; i += 256) {
    const int m = i >> 6, d = i & 63;
    float s = 0.f;
    for (int w = 0; w < 8; ++w) s += sCtx[(w * 16 + m) * DK_ + d];
    s *= sInv[m];
    ctx[((size_t)b * S_ + q0 + m) * D_ + h * DK_ + d] = f2bf(s);
  }
}

// ---------------------------------------------------------------------------
// Kernel 5: out = ctx @ fc_w^T + fc_b + residual   (fp32 result)
// ---------------------------------------------------------------------------
__global__ __launch_bounds__(128) void fc_kernel(
    const unsigned short* __restrict__ ctx,
    const unsigned short* __restrict__ fcw,
    const float* __restrict__ fcb,
    const float* __restrict__ xres,
    float* __restrict__ out) {
  const int lane = threadIdx.x & 31;
  const int wave = threadIdx.x >> 5;
  const bool hi = lane >= 16;
  const int ln16 = lane & 15;
  const int m0 = blockIdx.x * 64 + wave * 16;
  const int n0 = blockIdx.y * 64;

  v8f acc[4] = {};
  const unsigned short* arow = ctx + (size_t)(m0 + ln16) * D_;
  for (int kc = 0; kc < D_; kc += 32) {
    const v16bf a = load_a_frag(arow + kc, hi);
#pragma unroll
    for (int nt = 0; nt < 4; ++nt) {
      const v16bf b = load_b_frag(fcw + (size_t)(n0 + nt * 16 + ln16) * D_ + kc, hi);
      acc[nt] = wmma_bf16(a, b, acc[nt]);
    }
  }
#pragma unroll
  for (int nt = 0; nt < 4; ++nt) {
    const int n = n0 + nt * 16 + ln16;
#pragma unroll
    for (int v = 0; v < 8; ++v) {
      const int m = m0 + v + (hi ? 8 : 0);
      out[(size_t)m * D_ + n] = acc[nt][v] + fcb[n] + xres[(size_t)m * D_ + n];
    }
  }
}

// ---------------------------------------------------------------------------
// Host launcher
// ---------------------------------------------------------------------------
extern "C" void kernel_launch(void* const* d_in, const int* in_sizes, int n_in,
                              void* d_out, int out_size, void* d_ws, size_t ws_size,
                              hipStream_t stream) {
  (void)in_sizes; (void)n_in; (void)out_size; (void)ws_size;
  const float* x   = (const float*)d_in[0];
  const float* wq  = (const float*)d_in[1];
  const float* wk  = (const float*)d_in[2];
  const float* wv  = (const float*)d_in[3];
  const float* fcw = (const float*)d_in[4];
  const float* fcb = (const float*)d_in[5];
  const float* lng = (const float*)d_in[6];
  const float* lnb = (const float*)d_in[7];
  const unsigned char* mask = (const unsigned char*)d_in[8];

  float* out = (float*)d_out;                               // [B,S,D]
  float* attn_out = out + (size_t)NROWS * D_;               // [B,H,S,S]

  // workspace layout (u16 elements): 4x weights(1M) + xn(8M) + Q(8M) + K(8M)
  // + Vt(8M) + ctx(8M)  = 44M u16 = 88 MB
  unsigned short* w = (unsigned short*)d_ws;
  const size_t MU = 1024u * 1024u;
  unsigned short* wq_bf  = w;
  unsigned short* wk_bf  = w + 1 * MU;
  unsigned short* wv_bf  = w + 2 * MU;
  unsigned short* fcw_bf = w + 3 * MU;
  unsigned short* xn_bf  = w + 4 * MU;
  unsigned short* Qb     = w + 12 * MU;
  unsigned short* Kb     = w + 20 * MU;
  unsigned short* Vt     = w + 28 * MU;
  unsigned short* ctx    = w + 36 * MU;

  cvt_kernel<<<512, 256, 0, stream>>>(wq,  wq_bf,  D_ * D_);
  cvt_kernel<<<512, 256, 0, stream>>>(wk,  wk_bf,  D_ * D_);
  cvt_kernel<<<512, 256, 0, stream>>>(wv,  wv_bf,  D_ * D_);
  cvt_kernel<<<512, 256, 0, stream>>>(fcw, fcw_bf, D_ * D_);

  ln_kernel<<<NROWS, 256, 0, stream>>>(x, lng, lnb, xn_bf);

  proj_kernel<<<dim3(NROWS / 64, D_ / 64, 3), 128, 0, stream>>>(
      xn_bf, wq_bf, wk_bf, wv_bf, Qb, Kb, Vt);

  attn_kernel<<<dim3(S_ / 16, B_ * H_), 256, 0, stream>>>(
      Qb, Kb, Vt, mask, attn_out, ctx);

  fc_kernel<<<dim3(NROWS / 64, D_ / 64), 128, 0, stream>>>(
      ctx, fcw_bf, fcb, x, out);
}
